// NEFSProsodyPredictor_58025008169408
// MI455X (gfx1250) — matrix operand order repaired and checked by reading, not compiled, not run on hardware
//
#include <hip/hip_runtime.h>
#include <hip/hip_bf16.h>

#define NB 16
#define NS 512
#define IN_DIM 512
#define HID 256
#define NT 4096          // S * DUR_MAX

typedef __attribute__((ext_vector_type(16))) _Float16 v16h;
typedef __attribute__((ext_vector_type(8)))  float    v8f;

__device__ __forceinline__ v8f wmma16(v16h a, v16h b, v8f c) {
  // D = A(16x32 f16) * B(32x16 f16) + C(16x16 f32)
  return __builtin_amdgcn_wmma_f32_16x16x32_f16(false, a, false, b, (short)0, c, false, false);
}

// ---------------------------------------------------------------------------
// Packing: fp32 -> f16, activations into halo layout, weights into per-lane
// WMMA A-fragment order (contiguous k order: k = 16*(lane/16) + j per 32-k tile, same as B).
// ---------------------------------------------------------------------------
__global__ void k_pack_x(const float* __restrict__ x, _Float16* __restrict__ xh) {
  const int n = NB * NS * IN_DIM;
  for (int i = blockIdx.x * blockDim.x + threadIdx.x; i < n; i += gridDim.x * blockDim.x) {
    int b  = i / (NS * IN_DIM);
    int r  = i % (NS * IN_DIM);
    int s  = r / IN_DIM;
    int ch = r % IN_DIM;
    xh[((size_t)b * (NS + 2) + s + 1) * IN_DIM + ch] = (_Float16)x[i];
  }
}

__global__ void k_zero_halos(_Float16* __restrict__ xh, _Float16* __restrict__ h1) {
  int tid = blockIdx.x * blockDim.x + threadIdx.x;
  const int n1 = NB * 2 * IN_DIM;
  if (tid < n1) {
    int b = tid / (2 * IN_DIM); int j = tid % (2 * IN_DIM);
    int row = (j < IN_DIM) ? 0 : (NS + 1); int ch = j % IN_DIM;
    xh[((size_t)b * (NS + 2) + row) * IN_DIM + ch] = (_Float16)0.f;
  }
  const int n2 = NB * 2 * HID;
  if (tid < n2) {
    int b = tid / (2 * HID); int j = tid % (2 * HID);
    int row = (j < HID) ? 0 : (NS + 1); int ch = j % HID;
    h1[((size_t)b * (NS + 2) + row) * HID + ch] = (_Float16)0.f;
  }
}

// w: (HID out, Cin, 3 taps) fp32. K flattened as k = tap*Cin + ch.
// Packed element d = ((mt*nKt + kt)*32 + lane)*16 + j; each lane's 16
// contiguous halves form one A-fragment register group.
__global__ void k_pack_w(const float* __restrict__ w, _Float16* __restrict__ wp, int Cin) {
  const int nKt = (3 * Cin) / 32;
  const int n = HID * 3 * Cin;
  for (int i = blockIdx.x * blockDim.x + threadIdx.x; i < n; i += gridDim.x * blockDim.x) {
    int j  = i & 15;
    int L  = (i >> 4) & 31;
    int kt = (i >> 9) % nKt;
    int mt = i / (512 * nKt);
    int m  = mt * 16 + (L & 15);
    int grp = L >> 4;
    int kA = kt * 32 + 16 * grp + j;
    int ch = kA % Cin, tap = kA / Cin;
    wp[i] = (_Float16)w[((size_t)m * Cin + ch) * 3 + tap];
  }
}

// ---------------------------------------------------------------------------
// dp conv1 (512->256, k=3) + bias + ReLU + LayerNorm(ch) -> h1 (f16, halo)
// ---------------------------------------------------------------------------
__global__ __launch_bounds__(256) void k_dp_conv1(
    const _Float16* __restrict__ xh, const _Float16* __restrict__ wp,
    const float* __restrict__ bias, const float* __restrict__ g,
    const float* __restrict__ be, _Float16* __restrict__ out) {
  __shared__ __attribute__((aligned(32))) _Float16 lx[18 * IN_DIM];
  __shared__ float sB[HID], sG[HID], sBe[HID];
  __shared__ float redS[16], redQ[16];
  const int tid = threadIdx.x, lane = tid & 31, wave = tid >> 5;
  const int b = blockIdx.y, s0 = blockIdx.x * 16;

  { // stage rows s0-1 .. s0+16 (xh halo rows s0 .. s0+17)
    const uint4* src = (const uint4*)(xh + (size_t)(b * (NS + 2) + s0) * IN_DIM);
    uint4* dst = (uint4*)lx;
    __builtin_prefetch(src + 18 * IN_DIM / 8, 0, 1);   // next tile's rows
    for (int i = tid; i < 18 * IN_DIM / 8; i += 256) dst[i] = src[i];
  }
  if (tid < HID) { sB[tid] = bias[tid]; sG[tid] = g[tid]; sBe[tid] = be[tid]; }
  if (tid < 16)  { redS[tid] = 0.f; redQ[tid] = 0.f; }
  __syncthreads();

  const int mt0 = wave * 2, mt1 = mt0 + 1;
  const int grp = lane >> 4, n = lane & 15;
  const int nKt = 3 * IN_DIM / 32;   // 48
  // Linear-form bases: LDS B index = n*CIN + kt*32 + grp*16;
  // global A index = mt*nKt*512 + kt*512 + lane*16.
  const _Float16* lb  = lx + n * IN_DIM + grp * 16;
  const _Float16* wa0 = wp + (size_t)mt0 * nKt * 512 + lane * 16;
  const _Float16* wa1 = wp + (size_t)mt1 * nKt * 512 + lane * 16;
  v8f c0 = {}, c1 = {};
#pragma unroll
  for (int kt = 0; kt < nKt; ++kt) {
    v16h bf = *(const v16h*)(lb + kt * 32);
    v16h a0 = *(const v16h*)(wa0 + kt * 512);
    v16h a1 = *(const v16h*)(wa1 + kt * 512);
    c0 = wmma16(a0, bf, c0);
    c1 = wmma16(a1, bf, c1);
  }
  float v[16]; float ps = 0.f, pq = 0.f;
#pragma unroll
  for (int r = 0; r < 8; ++r) {
    int ch0 = mt0 * 16 + r + 8 * grp;
    int ch1 = mt1 * 16 + r + 8 * grp;
    float x0 = fmaxf(c0[r] + sB[ch0], 0.f);
    float x1 = fmaxf(c1[r] + sB[ch1], 0.f);
    v[r] = x0; v[8 + r] = x1;
    ps += x0 + x1; pq += x0 * x0 + x1 * x1;
  }
  atomicAdd(&redS[n], ps); atomicAdd(&redQ[n], pq);
  __syncthreads();
  float mean = redS[n] * (1.f / HID);
  float var  = redQ[n] * (1.f / HID) - mean * mean;
  float rstd = rsqrtf(var + 1e-5f);
  _Float16* orow = out + (size_t)(b * (NS + 2) + s0 + n + 1) * HID;
#pragma unroll
  for (int r = 0; r < 8; ++r) {
    int ch0 = mt0 * 16 + r + 8 * grp;
    int ch1 = mt1 * 16 + r + 8 * grp;
    orow[ch0] = (_Float16)((v[r]     - mean) * rstd * sG[ch0] + sBe[ch0]);
    orow[ch1] = (_Float16)((v[8 + r] - mean) * rstd * sG[ch1] + sBe[ch1]);
  }
}

// ---------------------------------------------------------------------------
// dp conv2 (256->256, k=3) + ReLU + LN, fused conv3 (1x1) + exp/max -> dur
// ---------------------------------------------------------------------------
__global__ __launch_bounds__(256) void k_dp_conv2(
    const _Float16* __restrict__ h1, const _Float16* __restrict__ wp,
    const float* __restrict__ bias, const float* __restrict__ g,
    const float* __restrict__ be, const float* __restrict__ w3,
    const float* __restrict__ b3, float* __restrict__ dur) {
  __shared__ __attribute__((aligned(32))) _Float16 lx[18 * HID];
  __shared__ float sB[HID], sG[HID], sBe[HID], sW3[HID];
  __shared__ float redS[16], redQ[16], red3[16];
  const int tid = threadIdx.x, lane = tid & 31, wave = tid >> 5;
  const int b = blockIdx.y, s0 = blockIdx.x * 16;

  {
    const uint4* src = (const uint4*)(h1 + (size_t)(b * (NS + 2) + s0) * HID);
    uint4* dst = (uint4*)lx;
    __builtin_prefetch(src + 18 * HID / 8, 0, 1);
    for (int i = tid; i < 18 * HID / 8; i += 256) dst[i] = src[i];
  }
  if (tid < HID) { sB[tid] = bias[tid]; sG[tid] = g[tid]; sBe[tid] = be[tid]; sW3[tid] = w3[tid]; }
  if (tid < 16)  { redS[tid] = 0.f; redQ[tid] = 0.f; red3[tid] = 0.f; }
  __syncthreads();

  const int mt0 = wave * 2, mt1 = mt0 + 1;
  const int grp = lane >> 4, n = lane & 15;
  const int nKt = 3 * HID / 32;     // 24
  const _Float16* lb  = lx + n * HID + grp * 16;
  const _Float16* wa0 = wp + (size_t)mt0 * nKt * 512 + lane * 16;
  const _Float16* wa1 = wp + (size_t)mt1 * nKt * 512 + lane * 16;
  v8f c0 = {}, c1 = {};
#pragma unroll
  for (int kt = 0; kt < nKt; ++kt) {
    v16h bf = *(const v16h*)(lb + kt * 32);
    v16h a0 = *(const v16h*)(wa0 + kt * 512);
    v16h a1 = *(const v16h*)(wa1 + kt * 512);
    c0 = wmma16(a0, bf, c0);
    c1 = wmma16(a1, bf, c1);
  }
  float v[16]; float ps = 0.f, pq = 0.f;
#pragma unroll
  for (int r = 0; r < 8; ++r) {
    int ch0 = mt0 * 16 + r + 8 * grp;
    int ch1 = mt1 * 16 + r + 8 * grp;
    float x0 = fmaxf(c0[r] + sB[ch0], 0.f);
    float x1 = fmaxf(c1[r] + sB[ch1], 0.f);
    v[r] = x0; v[8 + r] = x1;
    ps += x0 + x1; pq += x0 * x0 + x1 * x1;
  }
  atomicAdd(&redS[n], ps); atomicAdd(&redQ[n], pq);
  __syncthreads();
  float mean = redS[n] * (1.f / HID);
  float var  = redQ[n] * (1.f / HID) - mean * mean;
  float rstd = rsqrtf(var + 1e-5f);
  float pd = 0.f;
#pragma unroll
  for (int r = 0; r < 8; ++r) {
    int ch0 = mt0 * 16 + r + 8 * grp;
    int ch1 = mt1 * 16 + r + 8 * grp;
    float y0 = (v[r]     - mean) * rstd * sG[ch0] + sBe[ch0];
    float y1 = (v[8 + r] - mean) * rstd * sG[ch1] + sBe[ch1];
    pd += y0 * sW3[ch0] + y1 * sW3[ch1];
  }
  atomicAdd(&red3[n], pd);
  __syncthreads();
  if (tid < 16) {
    float ld = red3[tid] + b3[0];
    dur[(size_t)b * NS + s0 + tid] = fmaxf(expf(ld), 1.f);
  }
}

// ---------------------------------------------------------------------------
// Length regulator: cumsum + searchsorted gather
// ---------------------------------------------------------------------------
__global__ void k_cumsum(const int* __restrict__ d, int* __restrict__ cum) {
  int b = blockIdx.x * blockDim.x + threadIdx.x;
  if (b < NB) {
    int acc = 0;
    for (int s = 0; s < NS; ++s) { acc += d[b * NS + s]; cum[b * NS + s] = acc; }
  }
}

__global__ __launch_bounds__(128) void k_upsample(
    const float* __restrict__ x, const int* __restrict__ cum, float* __restrict__ up) {
  const int b = blockIdx.y, t = blockIdx.x;
  const int* c = cum + b * NS;
  const int total = c[NS - 1];
  float4 o;
  if (t < total) {
    int lo = 0, hi = NS;
    while (lo < hi) { int mid = (lo + hi) >> 1; if (c[mid] <= t) lo = mid + 1; else hi = mid; }
    int idx = lo < NS - 1 ? lo : NS - 1;
    o = ((const float4*)(x + ((size_t)b * NS + idx) * IN_DIM))[threadIdx.x];
  } else {
    o = make_float4(0.f, 0.f, 0.f, 0.f);
  }
  ((float4*)(up + ((size_t)b * NT + t) * IN_DIM))[threadIdx.x] = o;
}

// ---------------------------------------------------------------------------
// f0 conv1 (512->256, k=3) + ReLU + LN, fused f0 conv2 (1x1) -> predicted_f0
// Reads upsampled fp32 straight out of d_out (hot in L2), converts to f16 in LDS.
// ---------------------------------------------------------------------------
__global__ __launch_bounds__(256) void k_f0(
    const float* __restrict__ up, const _Float16* __restrict__ wp,
    const float* __restrict__ bias, const float* __restrict__ g,
    const float* __restrict__ be, const float* __restrict__ w2,
    const float* __restrict__ b2, float* __restrict__ f0) {
  __shared__ __attribute__((aligned(32))) _Float16 lx[18 * IN_DIM];
  __shared__ float sB[HID], sG[HID], sBe[HID], sW2[HID];
  __shared__ float redS[16], redQ[16], red2[16];
  const int tid = threadIdx.x, lane = tid & 31, wave = tid >> 5;
  const int b = blockIdx.y, t0 = blockIdx.x * 16;

  __builtin_prefetch(up + ((size_t)b * NT + t0 + 17) * IN_DIM, 0, 1);
  for (int i = tid; i < 18 * IN_DIM / 4; i += 256) {
    int row = i / (IN_DIM / 4);
    int c4  = i % (IN_DIM / 4);
    int t   = t0 - 1 + row;
    float4 val = (t >= 0 && t < NT)
        ? ((const float4*)(up + ((size_t)b * NT + t) * IN_DIM))[c4]
        : make_float4(0.f, 0.f, 0.f, 0.f);
    _Float16* p = &lx[row * IN_DIM + c4 * 4];
    p[0] = (_Float16)val.x; p[1] = (_Float16)val.y;
    p[2] = (_Float16)val.z; p[3] = (_Float16)val.w;
  }
  if (tid < HID) { sB[tid] = bias[tid]; sG[tid] = g[tid]; sBe[tid] = be[tid]; sW2[tid] = w2[tid]; }
  if (tid < 16)  { redS[tid] = 0.f; redQ[tid] = 0.f; red2[tid] = 0.f; }
  __syncthreads();

  const int mt0 = wave * 2, mt1 = mt0 + 1;
  const int grp = lane >> 4, n = lane & 15;
  const int nKt = 3 * IN_DIM / 32;  // 48
  const _Float16* lb  = lx + n * IN_DIM + grp * 16;
  const _Float16* wa0 = wp + (size_t)mt0 * nKt * 512 + lane * 16;
  const _Float16* wa1 = wp + (size_t)mt1 * nKt * 512 + lane * 16;
  v8f c0 = {}, c1 = {};
#pragma unroll
  for (int kt = 0; kt < nKt; ++kt) {
    v16h bf = *(const v16h*)(lb + kt * 32);
    v16h a0 = *(const v16h*)(wa0 + kt * 512);
    v16h a1 = *(const v16h*)(wa1 + kt * 512);
    c0 = wmma16(a0, bf, c0);
    c1 = wmma16(a1, bf, c1);
  }
  float v[16]; float ps = 0.f, pq = 0.f;
#pragma unroll
  for (int r = 0; r < 8; ++r) {
    int ch0 = mt0 * 16 + r + 8 * grp;
    int ch1 = mt1 * 16 + r + 8 * grp;
    float x0 = fmaxf(c0[r] + sB[ch0], 0.f);
    float x1 = fmaxf(c1[r] + sB[ch1], 0.f);
    v[r] = x0; v[8 + r] = x1;
    ps += x0 + x1; pq += x0 * x0 + x1 * x1;
  }
  atomicAdd(&redS[n], ps); atomicAdd(&redQ[n], pq);
  __syncthreads();
  float mean = redS[n] * (1.f / HID);
  float var  = redQ[n] * (1.f / HID) - mean * mean;
  float rstd = rsqrtf(var + 1e-5f);
  float pf = 0.f;
#pragma unroll
  for (int r = 0; r < 8; ++r) {
    int ch0 = mt0 * 16 + r + 8 * grp;
    int ch1 = mt1 * 16 + r + 8 * grp;
    float y0 = (v[r]     - mean) * rstd * sG[ch0] + sBe[ch0];
    float y1 = (v[8 + r] - mean) * rstd * sG[ch1] + sBe[ch1];
    pf += y0 * sW2[ch0] + y1 * sW2[ch1];
  }
  atomicAdd(&red2[n], pf);
  __syncthreads();
  if (tid < 16)
    f0[(size_t)b * NT + t0 + tid] = red2[tid] + b2[0];
}

// ---------------------------------------------------------------------------
extern "C" void kernel_launch(void* const* d_in, const int* in_sizes, int n_in,
                              void* d_out, int out_size, void* d_ws, size_t ws_size,
                              hipStream_t stream) {
  (void)in_sizes; (void)n_in; (void)out_size; (void)ws_size;
  const float* phon  = (const float*)d_in[0];
  const int*   tdur  = (const int*)d_in[1];
  const float* dp_w1 = (const float*)d_in[2];
  const float* dp_b1 = (const float*)d_in[3];
  const float* dp_g1 = (const float*)d_in[4];
  const float* dp_be1= (const float*)d_in[5];
  const float* dp_w2 = (const float*)d_in[6];
  const float* dp_b2 = (const float*)d_in[7];
  const float* dp_g2 = (const float*)d_in[8];
  const float* dp_be2= (const float*)d_in[9];
  const float* dp_w3 = (const float*)d_in[10];
  const float* dp_b3 = (const float*)d_in[11];
  const float* f0_w1 = (const float*)d_in[12];
  const float* f0_b1 = (const float*)d_in[13];
  const float* f0_g1 = (const float*)d_in[14];
  const float* f0_be1= (const float*)d_in[15];
  const float* f0_w2 = (const float*)d_in[16];
  const float* f0_b2 = (const float*)d_in[17];

  // Output layout: upsampled (B,T,512) | durations (B,S) | f0 (B,T)
  float* out = (float*)d_out;
  float* up  = out;
  float* dur = out + (size_t)NB * NT * IN_DIM;
  float* f0  = dur + (size_t)NB * NS;

  // Workspace carve-up (all offsets 256B aligned)
  char* ws = (char*)d_ws;
  _Float16* xh  = (_Float16*)(ws);                         // 16*514*512 f16
  _Float16* h1  = (_Float16*)(ws + 8421376);               // 16*514*256 f16
  _Float16* w1p = (_Float16*)(ws + 12632064);              // 256*1536 f16
  _Float16* w2p = (_Float16*)(ws + 13418496);              // 256*768  f16
  _Float16* f1p = (_Float16*)(ws + 13811712);              // 256*1536 f16
  int*      cum = (int*)     (ws + 14598144);              // 16*512 i32

  // 1) packing / init
  k_zero_halos<<<64, 256, 0, stream>>>(xh, h1);
  k_pack_x<<<2048, 256, 0, stream>>>(phon, xh);
  k_pack_w<<<512, 256, 0, stream>>>(dp_w1, w1p, IN_DIM);
  k_pack_w<<<512, 256, 0, stream>>>(dp_w2, w2p, HID);
  k_pack_w<<<512, 256, 0, stream>>>(f0_w1, f1p, IN_DIM);

  // 2) duration predictor (WMMA GEMMs, fused LN / 1x1 head)
  dim3 gS(NS / 16, NB);
  k_dp_conv1<<<gS, 256, 0, stream>>>(xh, w1p, dp_b1, dp_g1, dp_be1, h1);
  k_dp_conv2<<<gS, 256, 0, stream>>>(h1, w2p, dp_b2, dp_g2, dp_be2, dp_w3, dp_b3, dur);

  // 3) length regulator
  k_cumsum<<<1, 32, 0, stream>>>(tdur, cum);
  dim3 gU(NT, NB);
  k_upsample<<<gU, 128, 0, stream>>>(phon, cum, up);

  // 4) f0 predictor on upsampled frames (reads up out of L2, fused head)
  dim3 gT(NT / 16, NB);
  k_f0<<<gT, 256, 0, stream>>>(up, f1p, f0_b1, f0_g1, f0_be1, f0_w2, f0_b2, f0);
}
